// GT_80719615361178
// MI455X (gfx1250) — matrix operand order repaired and checked by reading, not compiled, not run on hardware
//
#include <hip/hip_runtime.h>

#define NNODES 50000
#define NEDGES 800000
#define HEADS  4
#define DHEAD  32
#define KDIM   128   // every GEMM contraction dim (DIN == H*DH == 128)

typedef __attribute__((ext_vector_type(16))) __bf16 v16bf;
typedef __attribute__((ext_vector_type(8)))  float  v8f;

union BF16x16 { v16bf v; uint4 u[2]; };

__device__ __forceinline__ unsigned short f32_to_bf16(float f) {
  unsigned u = __float_as_uint(f);
  unsigned r = 0x7FFFu + ((u >> 16) & 1u);   // round-to-nearest-even
  return (unsigned short)((u + r) >> 16);
}

__device__ __forceinline__ void atomic_max_f32(float* addr, float val) {
  // IEEE ordering trick: non-negative -> signed int max, negative -> unsigned min
  if (val >= 0.0f) atomicMax((int*)addr, __float_as_int(val));
  else             atomicMin((unsigned int*)addr, __float_as_uint(val));
}

// ---------------------------------------------------------------- pack X (f32 -> bf16)
__global__ void k_pack_x(const float* __restrict__ x, unsigned short* __restrict__ hbf, int n) {
  int gid = blockIdx.x * blockDim.x + threadIdx.x;
  if (gid < n) hbf[gid] = f32_to_bf16(x[gid]);
}

// ------------------------------------------- pack concat weights into B-fragment layout
// Packed element index = ((ct*4 + ks)*32 + lane)*16 + i
// maps to W element: K = ks*32 + (lane>=16 ? 8:0) + (i&7) + (i>=8 ? 16:0) ; col = ct*16 + (lane&15)
// Columns >= 384+dskip (layer-2 padding up to dstride) get zero weight/bias.
__global__ void k_pack_w(const float* __restrict__ Wq, const float* __restrict__ Wk,
                         const float* __restrict__ Wv, const float* __restrict__ Ws,
                         const float* __restrict__ bq, const float* __restrict__ bk,
                         const float* __restrict__ bv, const float* __restrict__ bs,
                         unsigned short* __restrict__ Wpk, float* __restrict__ bcat,
                         int dskip, int dstride) {
  int gid = blockIdx.x * blockDim.x + threadIdx.x;
  int total = KDIM * dstride;
  if (gid >= total) return;
  int i    = gid & 15;
  int lane = (gid >> 4) & 31;
  int ks   = (gid >> 9) & 3;
  int ct   = gid >> 11;
  int K    = ks * 32 + ((lane & 16) ? 8 : 0) + (i & 7) + ((i & 8) ? 16 : 0);
  int col  = ct * 16 + (lane & 15);
  float val;
  if      (col < 128)        val = Wq[K * 128 + col];
  else if (col < 256)        val = Wk[K * 128 + (col - 128)];
  else if (col < 384)        val = Wv[K * 128 + (col - 256)];
  else if (col < 384 + dskip) val = Ws[K * dskip + (col - 384)];
  else                       val = 0.0f;   // padded columns
  Wpk[gid] = f32_to_bf16(val);
  if (gid < dstride) {
    int c = gid; float b;
    if      (c < 128)         b = bq[c];
    else if (c < 256)         b = bk[c - 128];
    else if (c < 384)         b = bv[c - 256];
    else if (c < 384 + dskip) b = bs[c - 384];
    else                      b = 0.0f;
    bcat[c] = b;
  }
}

// ---------------------------------------------------------------- bf16 WMMA GEMM
// C[N, dstride] = A[N,128](bf16) * Wpk(bf16, frag layout) + bias
// One wave computes a 16x64 strip (4 column tiles): A fragment loaded once,
// reused across 4 accumulators -> 16 v_wmma per wave.
__global__ void k_gemm(const unsigned short* __restrict__ Abf,
                       const unsigned short* __restrict__ Bpk,
                       const float* __restrict__ bias,
                       float* __restrict__ C,
                       int dstride, int nCG, int nWaves) {
  int wave = blockIdx.x * (blockDim.x >> 5) + (threadIdx.x >> 5);
  if (wave >= nWaves) return;               // uniform per wave: EXEC stays all-ones
  int lane = threadIdx.x & 31;
  int cg = wave % nCG;                      // column group (4 tiles = 64 cols)
  int rt = wave / nCG;

  const uint4* Arow = (const uint4*)(Abf + (size_t)(rt * 16 + (lane & 15)) * KDIM);
  int kh = (lane & 16) ? 1 : 0;             // lanes 16-31 start at K=+8
  // B fragment base for column tile (cg*4): uint4 index = ct*256 + lane*2
  const uint4* Bp = (const uint4*)Bpk + ((size_t)(cg * 4) * 256 + (size_t)lane * 2);

  v8f acc0 = {}, acc1 = {}, acc2 = {}, acc3 = {};
#pragma unroll
  for (int ks = 0; ks < 4; ++ks) {
    BF16x16 a;
    a.u[0] = Arow[ks * 4 + kh];             // K = ks*32 + klo     .. +7
    a.u[1] = Arow[ks * 4 + kh + 2];         // K = ks*32 + klo + 16.. +23
    BF16x16 b0, b1, b2, b3;
    b0.u[0] = Bp[ks * 64 + 0];   b0.u[1] = Bp[ks * 64 + 1];
    b1.u[0] = Bp[256 + ks * 64]; b1.u[1] = Bp[256 + ks * 64 + 1];
    b2.u[0] = Bp[512 + ks * 64]; b2.u[1] = Bp[512 + ks * 64 + 1];
    b3.u[0] = Bp[768 + ks * 64]; b3.u[1] = Bp[768 + ks * 64 + 1];
    acc0 = __builtin_amdgcn_wmma_f32_16x16x32_bf16(false, a.v, false, b0.v, (short)0, acc0, false, false);
    acc1 = __builtin_amdgcn_wmma_f32_16x16x32_bf16(false, a.v, false, b1.v, (short)0, acc1, false, false);
    acc2 = __builtin_amdgcn_wmma_f32_16x16x32_bf16(false, a.v, false, b2.v, (short)0, acc2, false, false);
    acc3 = __builtin_amdgcn_wmma_f32_16x16x32_bf16(false, a.v, false, b3.v, (short)0, acc3, false, false);
  }

  int colb  = cg * 64 + (lane & 15);
  float bv0 = bias[colb];
  float bv1 = bias[colb + 16];
  float bv2 = bias[colb + 32];
  float bv3 = bias[colb + 48];
  int rbase = rt * 16 + ((lane & 16) ? 8 : 0);
#pragma unroll
  for (int r = 0; r < 8; ++r) {
    float* row = C + (size_t)(rbase + r) * dstride + colb;
    row[0]  = acc0[r] + bv0;
    row[16] = acc1[r] + bv1;
    row[32] = acc2[r] + bv2;
    row[48] = acc3[r] + bv3;
  }
}

// ---------------------------------------------------------------- per-layer init
__global__ void k_init(float* __restrict__ AGG, float* __restrict__ m,
                       float* __restrict__ s, int nAgg, int nMS) {
  int gid = blockIdx.x * blockDim.x + threadIdx.x;
  if (gid < nAgg) AGG[gid] = 0.0f;
  if (gid < nMS) { m[gid] = __int_as_float(0xFF800000); s[gid] = 0.0f; }
}

// ---------------------------------------------------------------- edge logits + seg-max
__global__ void k_logits(const float* __restrict__ QKVS,
                         const int* __restrict__ src, const int* __restrict__ dst,
                         float* __restrict__ logits, float* __restrict__ m, int dstride) {
  int gid = blockIdx.x * blockDim.x + threadIdx.x;
  if (gid >= NEDGES * HEADS) return;
  int e = gid >> 2, h = gid & 3;
  int di = dst[e], si = src[e];
  const float4* q = (const float4*)(QKVS + (size_t)di * dstride + h * DHEAD);
  const float4* k = (const float4*)(QKVS + (size_t)si * dstride + 128 + h * DHEAD);
  float acc = 0.0f;
#pragma unroll
  for (int j = 0; j < 8; ++j) {
    float4 a = q[j], b = k[j];
    acc += a.x * b.x + a.y * b.y + a.z * b.z + a.w * b.w;
  }
  acc *= 0.17677669529663687f;  // 1/sqrt(32)
  logits[gid] = acc;
  atomic_max_f32(&m[di * HEADS + h], acc);
}

// ---------------------------------------------------------------- exp + seg-sum
__global__ void k_expsum(float* __restrict__ logits, const float* __restrict__ m,
                         float* __restrict__ s, const int* __restrict__ dst) {
  int gid = blockIdx.x * blockDim.x + threadIdx.x;
  if (gid >= NEDGES * HEADS) return;
  int e = gid >> 2, h = gid & 3;
  int di = dst[e];
  float a = __expf(logits[gid] - m[di * HEADS + h]);
  logits[gid] = a;
  atomicAdd(&s[di * HEADS + h], a);
}

// ---------------------------------------------------------------- weighted message scatter
__global__ void k_message(const float* __restrict__ a, const float* __restrict__ s,
                          const float* __restrict__ QKVS,
                          const int* __restrict__ src, const int* __restrict__ dst,
                          float* __restrict__ AGG, int dstride) {
  long long gid = (long long)blockIdx.x * blockDim.x + threadIdx.x;
  if (gid >= (long long)NEDGES * 128) return;
  int e = (int)(gid >> 7);
  int t = (int)(gid & 127);
  int h = t >> 5;
  int di = dst[e], si = src[e];
  float alpha = a[e * HEADS + h] / s[di * HEADS + h];
  float val = QKVS[(size_t)si * dstride + 256 + t] * alpha;
  atomicAdd(&AGG[(size_t)di * 128 + t], val);
}

// ---------------------------------------------------------------- finalize layers 0,1
__global__ void k_final01(const float* __restrict__ QKVS, const float* __restrict__ AGG,
                          unsigned short* __restrict__ hbf, int dstride, int n) {
  int gid = blockIdx.x * blockDim.x + threadIdx.x;
  if (gid >= n) return;
  int node = gid >> 7, d = gid & 127;
  float o = QKVS[(size_t)node * dstride + 384 + d] + AGG[gid];
  o = fmaxf(o, 0.0f);                        // ReLU fused before next layer
  hbf[gid] = f32_to_bf16(o);
}

// ---------------------------------------------------------------- finalize layer 2 (head mean)
__global__ void k_final2(const float* __restrict__ QKVS, const float* __restrict__ AGG,
                         float* __restrict__ out, int dstride, int n) {
  int gid = blockIdx.x * blockDim.x + threadIdx.x;
  if (gid >= n) return;
  int node = gid >> 5, d = gid & 31;
  const float* ag = AGG + (size_t)node * 128;
  float mean = 0.25f * (ag[d] + ag[32 + d] + ag[64 + d] + ag[96 + d]);
  out[gid] = QKVS[(size_t)node * dstride + 384 + d] + mean;
}

extern "C" void kernel_launch(void* const* d_in, const int* in_sizes, int n_in,
                              void* d_out, int out_size, void* d_ws, size_t ws_size,
                              hipStream_t stream) {
  const float* x  = (const float*)d_in[0];
  const int*   ei = (const int*)d_in[1];
  const int*   src = ei;
  const int*   dst = ei + NEDGES;
  const float* W[3][8];
  for (int l = 0; l < 3; ++l)
    for (int j = 0; j < 8; ++j)
      W[l][j] = (const float*)d_in[2 + l * 8 + j];   // Wq,bq,Wk,bk,Wv,bv,Ws,bs

  // ---- workspace carve-out (≈156 MB) ----
  char* ws = (char*)d_ws;
  auto carve = [&](size_t bytes) {
    void* p = (void*)ws;
    ws += (bytes + 255) & ~(size_t)255;
    return p;
  };
  float*          QKVS   = (float*)         carve((size_t)NNODES * 512 * 4);
  float*          AGG    = (float*)         carve((size_t)NNODES * 128 * 4);
  unsigned short* hbf    = (unsigned short*)carve((size_t)NNODES * 128 * 2);
  float*          logits = (float*)         carve((size_t)NEDGES * HEADS * 4);
  float*          mbuf   = (float*)         carve((size_t)NNODES * HEADS * 4);
  float*          sbuf   = (float*)         carve((size_t)NNODES * HEADS * 4);
  unsigned short* Wpk    = (unsigned short*)carve((size_t)KDIM * 512 * 2);
  float*          bcat   = (float*)         carve((size_t)512 * 4);
  (void)ws_size; (void)in_sizes; (void)n_in; (void)out_size;

  const int B = 256;
  k_pack_x<<<(NNODES * 128 + B - 1) / B, B, 0, stream>>>(x, hbf, NNODES * 128);

  for (int l = 0; l < 3; ++l) {
    int dskip   = (l == 2) ? DHEAD : 128;
    int dstride = (l == 2) ? 448 : 512;  // padded to a multiple of 64 cols
    int nCG     = dstride / 64;          // 8, 8, 7 column groups
    int nWaves  = (NNODES / 16) * nCG;   // 50000 % 16 == 0

    k_pack_w<<<(KDIM * dstride + B - 1) / B, B, 0, stream>>>(
        W[l][0], W[l][2], W[l][4], W[l][6],
        W[l][1], W[l][3], W[l][5], W[l][7],
        Wpk, bcat, dskip, dstride);

    k_gemm<<<(nWaves + 3) / 4, 128, 0, stream>>>(hbf, Wpk, bcat, QKVS, dstride, nCG, nWaves);

    k_init<<<(NNODES * 128 + B - 1) / B, B, 0, stream>>>(
        AGG, mbuf, sbuf, NNODES * 128, NNODES * HEADS);

    int eh = NEDGES * HEADS;
    k_logits<<<(eh + B - 1) / B, B, 0, stream>>>(QKVS, src, dst, logits, mbuf, dstride);
    k_expsum<<<(eh + B - 1) / B, B, 0, stream>>>(logits, mbuf, sbuf, dst);

    long long msgN = (long long)NEDGES * 128;
    k_message<<<(unsigned)((msgN + B - 1) / B), B, 0, stream>>>(
        logits, sbuf, QKVS, src, dst, AGG, dstride);

    if (l < 2)
      k_final01<<<(NNODES * 128 + B - 1) / B, B, 0, stream>>>(QKVS, AGG, hbf, dstride, NNODES * 128);
    else
      k_final2<<<(NNODES * 32 + B - 1) / B, B, 0, stream>>>(QKVS, AGG, (float*)d_out, dstride, NNODES * 32);
  }
}
